// PillarResBackBone8xLargeKernel2D_40553081209465
// MI455X (gfx1250) — matrix operand order repaired and checked by reading, not compiled, not run on hardware
//
#include <hip/hip_runtime.h>

typedef __attribute__((ext_vector_type(16))) __bf16 v16bf;
typedef __attribute__((ext_vector_type(8)))  float  v8f;

__device__ __forceinline__ unsigned short f32_to_bf16(float a) {
  unsigned ua = __float_as_uint(a);
  ua = (ua + 0x7FFFu + ((ua >> 16) & 1u)) >> 16;   // round-to-nearest-even
  return (unsigned short)ua;
}
__device__ __forceinline__ float bf16_to_f32(unsigned short h) {
  return __uint_as_float(((unsigned)h) << 16);
}
__device__ __forceinline__ unsigned pack_bf16x2(float a, float b) {
  return (unsigned)f32_to_bf16(a) | ((unsigned)f32_to_bf16(b) << 16);
}

// ---------------------------------------------------------------------------
// Generic fused conv: implicit GEMM via v_wmma_f32_16x16x32_bf16.
// Activations live in bf16 (NHWC); each dword in memory = 2 consecutive
// channels = exactly one A-fragment K-pair, so the A fragment is 8 raw dword
// loads. Each wave: 16 output pixels x 32 output channels (2 accumulators,
// shared A). Outer loop over taps (address math once per tap), inner loop
// over 32-channel K chunks. Epilogue: relu((acc*scale+shift)*mask? + resid?),
// stored as bf16 (of32=0) or fp32 (of32=1, final layer -> d_out).
// ---------------------------------------------------------------------------
__global__ void conv_wmma_kernel(
    const unsigned short* __restrict__ in, const unsigned* __restrict__ wp,
    const float* __restrict__ scale, const float* __restrict__ shift,
    const float* __restrict__ mask, const unsigned short* __restrict__ resid,
    void* __restrict__ outp, int of32,
    int IH, int IW, int CI, int OH, int OW, int CO, int stride,
    int k1, int pad1, int dil1,
    int k2, int pad2, int dil2, int nch)
{
  const int lane = threadIdx.x & 31;
  const int wv   = threadIdx.x >> 5;
  const int wpb  = blockDim.x >> 5;
  const int cot0 = (blockIdx.y * wpb + wv) * 2;  // first 16-channel tile
  const int hf   = lane >> 4;
  const int lm   = lane & 15;

  const int hw  = OH * OW;
  const int gp  = blockIdx.x * 16;          // 16 consecutive pixels, same row
  const int b   = gp / hw;
  const int rem = gp - b * hw;
  const int oy  = rem / OW;
  const int ox0 = rem - oy * OW;
  const int axp = ox0 + lm;                 // this lane's A-row pixel x

  const int n1    = k1 * k1;
  const int ntaps = n1 + k2 * k2;

  v8f acc0 = {};
  v8f acc1 = {};
  // Pre-packed B streams: chunk kc lives at wp + (cot*nch + kc)*256 + lane*8
  const unsigned* bp0 = wp + (size_t)cot0 * nch * 256 + lane * 8;
  const unsigned* bp1 = bp0 + (size_t)nch * 256;

  for (int tap = 0; tap < ntaps; ++tap) {
    int kh, kw, pad, dil;
    if (tap < n1) { kh = tap / k1; kw = tap - kh * k1; pad = pad1; dil = dil1; }
    else { int t = tap - n1; kh = t / k2; kw = t - kh * k2; pad = pad2; dil = dil2; }
    const int iy = oy  * stride - pad + kh * dil;
    const int ix = axp * stride - pad + kw * dil;
    const bool inb = (iy >= 0) && (iy < IH) && (ix >= 0) && (ix < IW);
    const unsigned short* ap = in + ((b * IH + iy) * IW + ix) * CI;

    for (int c = 0; c < CI; c += 32) {
      // A fragment: lane m=lm, dword e holds K = c + hf*16 + e*2, +1
      union { v16bf v; unsigned u[8]; } Af;
      #pragma unroll
      for (int e = 0; e < 8; ++e) {
        const int cofs = c + hf * 16 + e * 2;
        Af.u[e] = inb ? *(const unsigned*)(ap + cofs) : 0u;
      }
      union { v16bf v; uint4 q[2]; } B0, B1;
      B0.q[0] = *(const uint4*)bp0;
      B0.q[1] = *(const uint4*)(bp0 + 4);
      B1.q[0] = *(const uint4*)bp1;
      B1.q[1] = *(const uint4*)(bp1 + 4);
      __builtin_prefetch(bp0 + 256, 0, 0);   // next chunk -> global_prefetch_b8
      __builtin_prefetch(bp1 + 256, 0, 0);

      acc0 = __builtin_amdgcn_wmma_f32_16x16x32_bf16(false, Af.v, false, B0.v,
                                                     (short)0, acc0, false, false);
      acc1 = __builtin_amdgcn_wmma_f32_16x16x32_bf16(false, Af.v, false, B1.v,
                                                     (short)0, acc1, false, false);
      bp0 += 256; bp1 += 256;
    }
  }

  // Epilogue. C/D layout: lane -> N=lm, VGPR r -> M = r + 8*hf
  const int   co0 = cot0 * 16 + lm;
  const int   co1 = co0 + 16;
  const float sc0 = scale[co0], sh0 = shift[co0];
  const float sc1 = scale[co1], sh1 = shift[co1];
  const int pixbase = b * hw + oy * OW + ox0;
  #pragma unroll
  for (int r = 0; r < 8; ++r) {
    const int pix  = pixbase + r + hf * 8;
    const float mv = mask ? mask[pix] : 1.f;
    float v0 = (acc0[r] * sc0 + sh0) * mv;
    float v1 = (acc1[r] * sc1 + sh1) * mv;
    if (resid) {
      v0 += bf16_to_f32(resid[pix * CO + co0]);
      v1 += bf16_to_f32(resid[pix * CO + co1]);
    }
    v0 = fmaxf(v0, 0.f);
    v1 = fmaxf(v1, 0.f);
    if (of32) {
      float* o = (float*)outp;
      o[pix * CO + co0] = v0;
      o[pix * CO + co1] = v1;
    } else {
      unsigned short* o = (unsigned short*)outp;
      o[pix * CO + co0] = f32_to_bf16(v0);
      o[pix * CO + co1] = f32_to_bf16(v1);
    }
  }
}

// ---------------------------------------------------------------------------
// Weight repack (HWIO fp32 -> bf16 B-fragment order).
// out idx = ((cot*nch + kc)*32 + lane)*8 + v ;  K = kc*32 + (lane>>4)*16 + 2v
// ---------------------------------------------------------------------------
__global__ void pack_weights_kernel(const float* __restrict__ w1,
                                    const float* __restrict__ w2,
                                    unsigned* __restrict__ out,
                                    int CI, int CO, int n1, int nch, int total)
{
  int idx = blockIdx.x * blockDim.x + threadIdx.x;
  if (idx >= total) return;
  const int v    = idx & 7;
  const int lane = (idx >> 3) & 31;
  const int rest = idx >> 8;               // cot*nch + kc
  const int kc   = rest % nch;
  const int cot  = rest / nch;
  const int K    = kc * 32 + ((lane >> 4) << 4) + v * 2;
  const int co   = cot * 16 + (lane & 15);
  const int tap  = K / CI;
  const int ci   = K - tap * CI;
  float a, b;
  if (tap < n1) {
    int base = (tap * CI + ci) * CO + co;
    a = w1[base]; b = w1[base + CO];
  } else {
    int base = ((tap - n1) * CI + ci) * CO + co;
    a = w2[base]; b = w2[base + CO];
  }
  out[idx] = pack_bf16x2(a, b);
}

// BN fold: scale = gamma*rsqrt(var+eps); shift = (bias1+bias2-mean)*scale+beta
__global__ void fold_bn_kernel(const float* __restrict__ g, const float* __restrict__ be,
                               const float* __restrict__ mn, const float* __restrict__ vr,
                               const float* __restrict__ b1, const float* __restrict__ b2,
                               float* __restrict__ sc, float* __restrict__ sh, int C)
{
  int c = blockIdx.x * blockDim.x + threadIdx.x;
  if (c >= C) return;
  float s = g[c] * rsqrtf(vr[c] + 1e-3f);
  float bias = (b1 ? b1[c] : 0.f) + (b2 ? b2[c] : 0.f);
  sc[c] = s;
  sh[c] = (bias - mn[c]) * s + be[c];
}

__global__ void i32_to_f32_kernel(const int* __restrict__ a, float* __restrict__ o, int n)
{
  int i = blockIdx.x * blockDim.x + threadIdx.x;
  if (i < n) o[i] = a[i] ? 1.f : 0.f;
}

// down_mask: any active input in 7x7/stride2/pad3 footprint
__global__ void down_mask_kernel(const float* __restrict__ mi, float* __restrict__ mo,
                                 int IH, int IW, int OH, int OW, int n)
{
  int i = blockIdx.x * blockDim.x + threadIdx.x;
  if (i >= n) return;
  int hw = OH * OW;
  int b = i / hw; int r = i - b * hw;
  int oy = r / OW; int ox = r - oy * OW;
  float v = 0.f;
  for (int kh = 0; kh < 7; ++kh) {
    int iy = oy * 2 - 3 + kh;
    if (iy < 0 || iy >= IH) continue;
    for (int kw = 0; kw < 7; ++kw) {
      int ix = ox * 2 - 3 + kw;
      if (ix < 0 || ix >= IW) continue;
      v += mi[(b * IH + iy) * IW + ix];
    }
  }
  mo[i] = (v > 0.f) ? 1.f : 0.f;
}

// h = x * mask, fp32 -> bf16
__global__ void mul_mask_kernel(const float* __restrict__ x, const float* __restrict__ m,
                                unsigned short* __restrict__ o, int n, int C)
{
  int i = blockIdx.x * blockDim.x + threadIdx.x;
  if (i < n) o[i] = f32_to_bf16(x[i] * m[i / C]);
}

// ---------------------------------------------------------------------------
// Host orchestration
// ---------------------------------------------------------------------------
struct BNp  { const float *g, *b, *m, *v; };
struct SGCp { const float *wk, *bk, *w3, *b3; };

static void launch_conv(hipStream_t st, const unsigned short* in, const unsigned* wp,
                        const float* sc, const float* sh, const float* mask,
                        const unsigned short* resid, void* out, int of32,
                        int Bn, int IH, int IW, int CI, int OH, int OW, int CO,
                        int stride, int k1, int pad1, int dil1,
                        int k2, int pad2, int dil2)
{
  int waves = CO / 32; if (waves > 8) waves = 8; if (waves < 1) waves = 1;
  dim3 blk(waves * 32, 1, 1);
  dim3 grd((unsigned)(Bn * OH * OW / 16), (unsigned)(CO / (32 * waves)), 1);
  int nch = (k1 * k1 + k2 * k2) * CI / 32;
  conv_wmma_kernel<<<grd, blk, 0, st>>>(in, wp, sc, sh, mask, resid, out, of32,
      IH, IW, CI, OH, OW, CO, stride, k1, pad1, dil1, k2, pad2, dil2, nch);
}

extern "C" void kernel_launch(void* const* d_in, const int* in_sizes, int n_in,
                              void* d_out, int out_size, void* d_ws, size_t ws_size,
                              hipStream_t stream)
{
  (void)in_sizes; (void)n_in; (void)out_size; (void)ws_size;
  int ii = 0;
  auto F = [&]() -> const float* { return (const float*)d_in[ii++]; };

  const float* x      = F();
  const int*   mask_i = (const int*)d_in[ii++];

  auto getBN  = [&]() { BNp p;  p.g = F(); p.b = F(); p.m = F(); p.v = F(); return p; };
  auto getSGC = [&]() { SGCp p; p.wk = F(); p.bk = F(); p.w3 = F(); p.b3 = F(); return p; };

  SGCp c1c[4], c2c[4]; BNp c1bn1[4], c1bn2[4];
  for (int i = 0; i < 4; ++i) { c1c[i] = getSGC(); c1bn1[i] = getBN(); c2c[i] = getSGC(); c1bn2[i] = getBN(); }
  const float* d2w = F(); BNp d2bn = getBN();
  const float *r2w1[2], *r2b1[2], *r2w2[2], *r2b2[2]; BNp r2bn1[2], r2bn2[2];
  for (int i = 0; i < 2; ++i) { r2w1[i]=F(); r2b1[i]=F(); r2bn1[i]=getBN(); r2w2[i]=F(); r2b2[i]=F(); r2bn2[i]=getBN(); }
  const float* d3w = F(); BNp d3bn = getBN();
  const float *r3w1[2], *r3b1[2], *r3w2[2], *r3b2[2]; BNp r3bn1[2], r3bn2[2];
  for (int i = 0; i < 2; ++i) { r3w1[i]=F(); r3b1[i]=F(); r3bn1[i]=getBN(); r3w2[i]=F(); r3b2[i]=F(); r3bn2[i]=getBN(); }
  const float* d4w = F(); BNp d4bn = getBN();
  const float *r4w1, *r4b1, *r4w2, *r4b2; BNp r4bn1, r4bn2;
  r4w1=F(); r4b1=F(); r4bn1=getBN(); r4w2=F(); r4b2=F(); r4bn2=getBN();
  const float* c5w0 = F(); BNp c5bn0 = getBN();
  const float* c5w1 = F(); BNp c5bn1 = getBN();
  const float* c5w2 = F(); BNp c5bn2 = getBN();

  // workspace bump allocator (256B aligned)
  char* wsb = (char*)d_ws; size_t off = 0;
  auto alloc = [&](size_t bytes) -> void* {
    void* p = wsb + off;
    off = (off + bytes + 255) & ~(size_t)255;
    return p;
  };

  const int B = 2;
  unsigned short* bufA = (unsigned short*)alloc((size_t)B * 512 * 512 * 32 * 2);
  unsigned short* bufB = (unsigned short*)alloc((size_t)B * 512 * 512 * 32 * 2);
  float* m1 = (float*)alloc((size_t)B * 512 * 512 * 4);
  float* m2 = (float*)alloc((size_t)B * 256 * 256 * 4);
  float* m3 = (float*)alloc((size_t)B * 128 * 128 * 4);
  float* m4 = (float*)alloc((size_t)B * 64 * 64 * 4);

  auto fold = [&](BNp bn, const float* b1, const float* b2, int C,
                  float** sc, float** sh) {
    *sc = (float*)alloc((size_t)C * 4);
    *sh = (float*)alloc((size_t)C * 4);
    fold_bn_kernel<<<(C + 63) / 64, 64, 0, stream>>>(bn.g, bn.b, bn.m, bn.v, b1, b2, *sc, *sh, C);
  };
  auto pack = [&](const float* w1, const float* w2, int CI, int CO, int k1, int k2) -> unsigned* {
    int n1 = k1 * k1;
    int nch = ((n1 + k2 * k2) * CI) / 32;
    int total = (CO / 16) * nch * 256;
    unsigned* wp = (unsigned*)alloc((size_t)total * 4);
    pack_weights_kernel<<<(total + 255) / 256, 256, 0, stream>>>(w1, w2, wp, CI, CO, n1, nch, total);
    return wp;
  };

  // masks + initial h = x * m1 (bf16)
  { int n = B * 512 * 512;
    i32_to_f32_kernel<<<(n + 255) / 256, 256, 0, stream>>>(mask_i, m1, n); }
  { int n = B * 256 * 256;
    down_mask_kernel<<<(n + 255) / 256, 256, 0, stream>>>(m1, m2, 512, 512, 256, 256, n); }
  { int n = B * 128 * 128;
    down_mask_kernel<<<(n + 255) / 256, 256, 0, stream>>>(m2, m3, 256, 256, 128, 128, n); }
  { int n = B * 64 * 64;
    down_mask_kernel<<<(n + 255) / 256, 256, 0, stream>>>(m3, m4, 128, 128, 64, 64, n); }
  { int n = B * 512 * 512 * 32;
    mul_mask_kernel<<<(n + 255) / 256, 256, 0, stream>>>(x, m1, bufA, n, 32); }

  // conv1: 4x SparseBasicBlock(32, k=7 sgc)  [bufA -> bufA]
  for (int i = 0; i < 4; ++i) {
    float *s1, *t1, *s2, *t2;
    fold(c1bn1[i], c1c[i].bk, c1c[i].b3, 32, &s1, &t1);
    fold(c1bn2[i], c2c[i].bk, c2c[i].b3, 32, &s2, &t2);
    unsigned* wp1 = pack(c1c[i].wk, c1c[i].w3, 32, 32, 7, 3);
    unsigned* wp2 = pack(c2c[i].wk, c2c[i].w3, 32, 32, 7, 3);
    launch_conv(stream, bufA, wp1, s1, t1, m1, nullptr, bufB, 0, B, 512,512,32, 512,512,32, 1, 7,3,1, 3,2,2);
    launch_conv(stream, bufB, wp2, s2, t2, m1, bufA,    bufA, 0, B, 512,512,32, 512,512,32, 1, 7,3,1, 3,2,2);
  }

  auto res_block = [&](const float* w1, const float* b1, BNp bn1,
                       const float* w2, const float* b2, BNp bn2,
                       unsigned short* cur, unsigned short* tmp,
                       int C, int HH, int WW, const float* mk) {
    float *s1, *t1, *s2, *t2;
    fold(bn1, b1, nullptr, C, &s1, &t1);
    fold(bn2, b2, nullptr, C, &s2, &t2);
    unsigned* p1 = pack(w1, nullptr, C, C, 3, 0);
    unsigned* p2 = pack(w2, nullptr, C, C, 3, 0);
    launch_conv(stream, cur, p1, s1, t1, mk, nullptr, tmp, 0, B, HH,WW,C, HH,WW,C, 1, 3,1,1, 0,0,1);
    launch_conv(stream, tmp, p2, s2, t2, mk, cur,     cur, 0, B, HH,WW,C, HH,WW,C, 1, 3,1,1, 0,0,1);
  };

  // down2 + res2  [bufA -> bufB, stays bufB]
  { float *s, *t; fold(d2bn, nullptr, nullptr, 64, &s, &t);
    unsigned* wp = pack(d2w, nullptr, 32, 64, 7, 0);
    launch_conv(stream, bufA, wp, s, t, m2, nullptr, bufB, 0, B, 512,512,32, 256,256,64, 2, 7,3,1, 0,0,1); }
  for (int i = 0; i < 2; ++i)
    res_block(r2w1[i], r2b1[i], r2bn1[i], r2w2[i], r2b2[i], r2bn2[i], bufB, bufA, 64, 256, 256, m2);

  // down3 + res3  [bufB -> bufA, stays bufA]
  { float *s, *t; fold(d3bn, nullptr, nullptr, 128, &s, &t);
    unsigned* wp = pack(d3w, nullptr, 64, 128, 7, 0);
    launch_conv(stream, bufB, wp, s, t, m3, nullptr, bufA, 0, B, 256,256,64, 128,128,128, 2, 7,3,1, 0,0,1); }
  for (int i = 0; i < 2; ++i)
    res_block(r3w1[i], r3b1[i], r3bn1[i], r3w2[i], r3b2[i], r3bn2[i], bufA, bufB, 128, 128, 128, m3);

  // down4 + res4  [bufA -> bufB, stays bufB]
  { float *s, *t; fold(d4bn, nullptr, nullptr, 256, &s, &t);
    unsigned* wp = pack(d4w, nullptr, 128, 256, 7, 0);
    launch_conv(stream, bufA, wp, s, t, m4, nullptr, bufB, 0, B, 128,128,128, 64,64,256, 2, 7,3,1, 0,0,1); }
  res_block(r4w1, r4b1, r4bn1, r4w2, r4b2, r4bn2, bufB, bufA, 256, 64, 64, m4);

  // conv5 (dense): y = relu(bn0(conv7s2))  [bufB -> bufA]
  { float *s, *t; fold(c5bn0, nullptr, nullptr, 256, &s, &t);
    unsigned* wp = pack(c5w0, nullptr, 256, 256, 7, 0);
    launch_conv(stream, bufB, wp, s, t, nullptr, nullptr, bufA, 0, B, 64,64,256, 32,32,256, 2, 7,3,1, 0,0,1); }
  // z1 = relu(bn1(conv3(y)))  [bufA -> bufB]
  { float *s, *t; fold(c5bn1, nullptr, nullptr, 256, &s, &t);
    unsigned* wp = pack(c5w1, nullptr, 256, 256, 3, 0);
    launch_conv(stream, bufA, wp, s, t, nullptr, nullptr, bufB, 0, B, 32,32,256, 32,32,256, 1, 3,1,1, 0,0,1); }
  // out = relu(bn2(conv3(z1)) + y)  [bufB -> d_out, fp32]
  { float *s, *t; fold(c5bn2, nullptr, nullptr, 256, &s, &t);
    unsigned* wp = pack(c5w2, nullptr, 256, 256, 3, 0);
    launch_conv(stream, bufB, wp, s, t, nullptr, bufA, d_out, 1, B, 32,32,256, 32,32,256, 1, 3,1,1, 0,0,1); }
}
